// MultiHeadAttention_33801392619605
// MI455X (gfx1250) — compile-verified, matrix operand order checked by reading, not run on hardware
//
#include <hip/hip_runtime.h>
#include <cstdint>

// ---------------------------------------------------------------------------
// Types for gfx1250 WMMA (wave32)
// ---------------------------------------------------------------------------
typedef __bf16 bf16;
typedef __attribute__((ext_vector_type(16))) __bf16 v16bf;
typedef __attribute__((ext_vector_type(8)))  __bf16 v8bf;
typedef __attribute__((ext_vector_type(4)))  __bf16 v4bf;
typedef __attribute__((ext_vector_type(8)))  float  v8f;
typedef __attribute__((ext_vector_type(4)))  float  v4f;

#define WMMA_BF16(a, b, c) \
  __builtin_amdgcn_wmma_f32_16x16x32_bf16(false, (a), false, (b), (short)0, (c), false, false)

// Two 16B loads -> one 16x32 bf16 fragment (VGPR j/2; K = half*8+j / 16+half*8+j-8).
static __device__ __forceinline__ v16bf frag_cat(v8bf lo, v8bf hi) {
  v16bf r;
#pragma unroll
  for (int i = 0; i < 8; ++i) { r[i] = lo[i]; r[i + 8] = hi[i]; }
  return r;
}

// ---------------------------------------------------------------------------
// CDNA5 async global->LDS copies (ASYNCcnt path) + partial waits.
// ASYNCcnt retires in order, so "<= N" waits give copy/compute overlap.
// ---------------------------------------------------------------------------
static __device__ __forceinline__ void async_b128(void* lds_dst, const void* gsrc) {
  unsigned off = (unsigned)(uintptr_t)lds_dst;   // low 32 bits of flat addr = LDS offset
  asm volatile("global_load_async_to_lds_b128 %0, %1, off"
               :: "v"(off), "v"(gsrc) : "memory");
}
static __device__ __forceinline__ void wait_async0() {
  asm volatile("s_wait_asynccnt 0x0" ::: "memory");
}
static __device__ __forceinline__ void wait_async2() {
  asm volatile("s_wait_asynccnt 0x2" ::: "memory");
}
static __device__ __forceinline__ void wait_async4() {
  asm volatile("s_wait_asynccnt 0x4" ::: "memory");
}

// ---------------------------------------------------------------------------
// Single-instruction DPP16 ROW_XMASK reductions (no canonicalize, no mov_dpp):
//   v_max_num_f32_dpp d, s(xor), s   /   v_add_f32_dpp d, s(xor), s
// Reduction domain = 16-lane half, which is exactly the WMMA C-layout row.
// ---------------------------------------------------------------------------
static __device__ __forceinline__ float fmax2(float a, float b) {
  float r;
  asm("v_max_num_f32_e32 %0, %1, %2" : "=v"(r) : "v"(a), "v"(b));
  return r;
}
#define DEF_DPP_RED(NAME, OPSTR, MASK)                                        \
  static __device__ __forceinline__ float NAME(float x) {                     \
    float r;                                                                  \
    asm(OPSTR " %0, %1, %1 row_xmask:" #MASK                                  \
        " row_mask:0xf bank_mask:0xf bound_ctrl:1"                            \
        : "=v"(r) : "v"(x));                                                  \
    return r;                                                                 \
  }
DEF_DPP_RED(dmax1, "v_max_num_f32_dpp", 1)
DEF_DPP_RED(dmax2, "v_max_num_f32_dpp", 2)
DEF_DPP_RED(dmax4, "v_max_num_f32_dpp", 4)
DEF_DPP_RED(dmax8, "v_max_num_f32_dpp", 8)
DEF_DPP_RED(dadd1, "v_add_f32_dpp", 1)
DEF_DPP_RED(dadd2, "v_add_f32_dpp", 2)
DEF_DPP_RED(dadd4, "v_add_f32_dpp", 4)
DEF_DPP_RED(dadd8, "v_add_f32_dpp", 8)

static __device__ __forceinline__ float rmax16(float x) {
  x = dmax1(x); x = dmax2(x); x = dmax4(x); x = dmax8(x);
  return x;
}
static __device__ __forceinline__ float rsum16(float x) {
  x = dadd1(x); x = dadd2(x); x = dadd4(x); x = dadd8(x);
  return x;
}

// ---------------------------------------------------------------------------
// fp32 -> bf16 straight conversion (X)
// ---------------------------------------------------------------------------
__global__ void f32_to_bf16_kernel(const float* __restrict__ in,
                                   bf16* __restrict__ out, int n) {
  int i = (blockIdx.x * blockDim.x + threadIdx.x) * 4;
  if (i + 3 < n) {
    v4f x = *(const v4f*)(in + i);
    v4bf y;
#pragma unroll
    for (int k = 0; k < 4; ++k) y[k] = (bf16)x[k];
    *(v4bf*)(out + i) = y;
  } else {
    for (; i < n; ++i) out[i] = (bf16)in[i];
  }
}

// ---------------------------------------------------------------------------
// fp32 [K][N] -> bf16 [N][K] transpose-convert (weights pre-pass).
// ---------------------------------------------------------------------------
__global__ void transpose_f32_to_bf16_kernel(const float* __restrict__ in,
                                             bf16* __restrict__ out,
                                             int K, int N) {
  __shared__ float t[32][33];
  const int n0 = blockIdx.x * 32, k0 = blockIdx.y * 32;
  const int tx = threadIdx.x, ty = threadIdx.y;   // block (32, 8)
#pragma unroll
  for (int i = ty; i < 32; i += 8)
    t[i][tx] = in[(size_t)(k0 + i) * N + n0 + tx];
  __syncthreads();
#pragma unroll
  for (int i = ty; i < 32; i += 8)
    out[(size_t)(n0 + i) * K + k0 + tx] = (bf16)t[tx][i];
}

// ---------------------------------------------------------------------------
// bf16 WMMA GEMM: 128x128 tile / 256-thread WG, 8 waves = 4M x 2N, each wave
// 32x64 (8 WMMAs per 32-k step). Async double-buffered LDS staging.
//   MODE 0: QKV epilogue -> bias; Q(*1/8)/K as [b,h,l,d]; V as [b,h,d,l]
//   MODE 1: FC  epilogue -> bias, f32 row-major output
// ---------------------------------------------------------------------------
template <int MODE>
__global__ __launch_bounds__(256) void gemm_bf16_kernel(
    const bf16* __restrict__ A,    // [M][K] row-major
    const bf16* __restrict__ Bt,   // [N][K] row-major (pre-transposed weights)
    const float* __restrict__ bias,
    float* __restrict__ Cout,                       // MODE 1
    bf16* __restrict__ Qb, bf16* __restrict__ Kb,   // MODE 0
    bf16* __restrict__ Vb,
    int Ndim, int Kdim) {
  __shared__ alignas(16) bf16 As[2][128][32];   // [m][k]
  __shared__ alignas(16) bf16 Bs[2][128][32];   // [n][k]

  const int tid  = threadIdx.x;
  const int lane = tid & 31;
  const int wave = tid >> 5;
  const int wm   = wave >> 1;        // 0..3 -> 32-row slice
  const int wn   = wave & 1;         // 0..1 -> 64-col slice
  const int col  = lane & 15;
  const int hf   = lane >> 4;

  const int tileM = blockIdx.y * 128;
  const int tileN = blockIdx.x * 128;

  const int crow = tid >> 1;         // 0..127 (copy row)
  const int ccol = (tid & 1) * 16;   // 0 / 16 (copy col base)

  const bf16* Aglob = A  + (size_t)(tileM + crow) * Kdim + ccol;
  const bf16* Bglob = Bt + (size_t)(tileN + crow) * Kdim + ccol;

  v8f z = {};
  v8f acc[2][4];
#pragma unroll
  for (int a = 0; a < 2; ++a)
#pragma unroll
    for (int s = 0; s < 4; ++s) acc[a][s] = z;

  const int nk = Kdim >> 5;
  async_b128(&As[0][crow][ccol],     Aglob);
  async_b128(&As[0][crow][ccol + 8], Aglob + 8);
  async_b128(&Bs[0][crow][ccol],     Bglob);
  async_b128(&Bs[0][crow][ccol + 8], Bglob + 8);

  for (int it = 0; it < nk; ++it) {
    wait_async0();
    __syncthreads();                 // tile[cur] visible; prev reads retired
    const int cur = it & 1;
    if (it + 1 < nk) {               // overlap next-stage DMA with WMMAs
      const int nxt = cur ^ 1;
      const bf16* an = Aglob + (it + 1) * 32;
      const bf16* bn = Bglob + (it + 1) * 32;
      async_b128(&As[nxt][crow][ccol],     an);
      async_b128(&As[nxt][crow][ccol + 8], an + 8);
      async_b128(&Bs[nxt][crow][ccol],     bn);
      async_b128(&Bs[nxt][crow][ccol + 8], bn + 8);
    }
    v16bf bfr[4];
#pragma unroll
    for (int s = 0; s < 4; ++s) {
      const int n = wn * 64 + s * 16 + col;
      bfr[s] = frag_cat(*(const v8bf*)&Bs[cur][n][hf * 8],
                        *(const v8bf*)&Bs[cur][n][16 + hf * 8]);
    }
#pragma unroll
    for (int a = 0; a < 2; ++a) {
      const int m = wm * 32 + a * 16 + col;
      v16bf af = frag_cat(*(const v8bf*)&As[cur][m][hf * 8],
                          *(const v8bf*)&As[cur][m][16 + hf * 8]);
#pragma unroll
      for (int s = 0; s < 4; ++s) acc[a][s] = WMMA_BF16(af, bfr[s], acc[a][s]);
    }
  }

  // Epilogue. D layout: n = lane%16, m = (lane/16)*8 + r.
#pragma unroll
  for (int a = 0; a < 2; ++a)
#pragma unroll
    for (int s = 0; s < 4; ++s)
#pragma unroll
      for (int r = 0; r < 8; ++r) {
        int grow = tileM + wm * 32 + a * 16 + hf * 8 + r;
        int gcol = tileN + wn * 64 + s * 16 + col;
        float v = acc[a][s][r] + bias[gcol];
        if (MODE == 1) {
          Cout[(size_t)grow * Ndim + gcol] = v;
        } else {
          int bi = grow >> 11, li = grow & 2047;         // L = 2048
          int h = gcol / 192, rr = gcol % 192;
          if (rr < 64)
            Qb[(((size_t)bi * 16 + h) * 2048 + li) * 64 + rr] = (bf16)(v * 0.125f);
          else if (rr < 128)
            Kb[(((size_t)bi * 16 + h) * 2048 + li) * 64 + (rr - 64)] = (bf16)v;
          else  // V stored [b,h,d,l] so attention tiles are copy-contiguous
            Vb[(((size_t)bi * 16 + h) * 64 + (rr - 128)) * 2048 + li] = (bf16)v;
        }
      }
}

// ---------------------------------------------------------------------------
// Flash attention: WG (256 thr / 8 waves) = 128 q-rows of one (b,h).
// 64-key tiles, 16 WMMAs per tile per wave. Async copies issued K-first so
// partial ASYNCcnt waits overlap the V DMA with S-WMMAs + softmax.
// ---------------------------------------------------------------------------
__global__ __launch_bounds__(256) void attn_kernel(
    const bf16* __restrict__ Qb, const bf16* __restrict__ Kb,
    const bf16* __restrict__ Vt,   // [b,h,d,l]
    bf16* __restrict__ Ob) {
  __shared__ alignas(16) bf16 Ks[2][64][64];    // [k][d] row-major
  __shared__ alignas(16) bf16 Vs[2][64][64];    // [d][k] (transposed in memory)
  __shared__ alignas(16) bf16 Pb[8][16][64];    // per-wave P C->A relayout bounce

  const int tid  = threadIdx.x;
  const int lane = tid & 31;
  const int w    = tid >> 5;
  const int col  = lane & 15;
  const int hf   = lane >> 4;

  const int bh    = blockIdx.y;
  const int b     = bh >> 4;
  const int h     = bh & 15;
  const int qtile = blockIdx.x;                 // 16 tiles of 128 q-rows

  const bf16* Qrow = Qb + (((size_t)bh * 2048) + qtile * 128 + w * 16 + col) * 64;
  v16bf qf0 = frag_cat(*(const v8bf*)(Qrow + hf * 8),      *(const v8bf*)(Qrow + 16 + hf * 8));
  v16bf qf1 = frag_cat(*(const v8bf*)(Qrow + 32 + hf * 8), *(const v8bf*)(Qrow + 48 + hf * 8));

  v8f z = {};
  v8f Of[4];
#pragma unroll
  for (int f = 0; f < 4; ++f) Of[f] = z;
  float rowmax[8], rowsum[8];
#pragma unroll
  for (int r = 0; r < 8; ++r) { rowmax[r] = -1e30f; rowsum[r] = 0.f; }

  const bf16* Kbase = Kb + (size_t)bh * 2048 * 64;
  const bf16* Vbase = Vt + (size_t)bh * 64 * 2048;

  const int trow = tid >> 2;              // 0..63
  const int tcol = (tid & 3) * 16;        // 0/16/32/48

  // Prologue (K issued before V -> in-order ASYNCcnt retirement).
  async_b128(&Ks[0][trow][tcol],     Kbase + (size_t)trow * 64 + tcol);
  async_b128(&Ks[0][trow][tcol + 8], Kbase + (size_t)trow * 64 + tcol + 8);
  async_b128(&Vs[0][trow][tcol],     Vbase + (size_t)trow * 2048 + tcol);
  async_b128(&Vs[0][trow][tcol + 8], Vbase + (size_t)trow * 2048 + tcol + 8);

  for (int j = 0; j < 32; ++j) {          // 32 tiles of 64 keys
    wait_async2();                        // K[cur] done; V[cur] may still fly
    __syncthreads();
    const int cur = j & 1;
    if (j + 1 < 32) {
      const int nxt = cur ^ 1;
      const bf16* kn = Kbase + ((size_t)(j + 1) * 64 + trow) * 64 + tcol;
      const bf16* vn = Vbase + (size_t)trow * 2048 + (j + 1) * 64 + tcol;
      async_b128(&Ks[nxt][trow][tcol],     kn);
      async_b128(&Ks[nxt][trow][tcol + 8], kn + 8);
      async_b128(&Vs[nxt][trow][tcol],     vn);
      async_b128(&Vs[nxt][trow][tcol + 8], vn + 8);
    }

    // S = Q*K^T (16x64): 4 n-subtiles x 2 k-steps.
    v8f S[4];
#pragma unroll
    for (int s = 0; s < 4; ++s) {
      v16bf kf0 = frag_cat(*(const v8bf*)&Ks[cur][s * 16 + col][hf * 8],
                           *(const v8bf*)&Ks[cur][s * 16 + col][16 + hf * 8]);
      v16bf kf1 = frag_cat(*(const v8bf*)&Ks[cur][s * 16 + col][32 + hf * 8],
                           *(const v8bf*)&Ks[cur][s * 16 + col][48 + hf * 8]);
      S[s] = WMMA_BF16(qf0, kf0, z);
      S[s] = WMMA_BF16(qf1, kf1, S[s]);
    }

    // Online softmax with fused-DPP row reductions.
#pragma unroll
    for (int r = 0; r < 8; ++r) {
      float m = fmax2(fmax2(S[0][r], S[1][r]), fmax2(S[2][r], S[3][r]));
      m = rmax16(m);
      float nm = fmax2(rowmax[r], m);
      float sc = __expf(rowmax[r] - nm);
      float p0 = __expf(S[0][r] - nm);
      float p1 = __expf(S[1][r] - nm);
      float p2 = __expf(S[2][r] - nm);
      float p3 = __expf(S[3][r] - nm);
      float ps = rsum16((p0 + p1) + (p2 + p3));
      rowsum[r] = rowsum[r] * sc + ps;
      rowmax[r] = nm;
      Pb[w][hf * 8 + r][col]      = (bf16)p0;
      Pb[w][hf * 8 + r][col + 16] = (bf16)p1;
      Pb[w][hf * 8 + r][col + 32] = (bf16)p2;
      Pb[w][hf * 8 + r][col + 48] = (bf16)p3;
#pragma unroll
      for (int f = 0; f < 4; ++f) Of[f][r] *= sc;
    }

    // V[cur] must be resident before P*V; partial wait keeps next-stage flying.
    if (j + 1 < 32) wait_async4(); else wait_async0();
    __syncthreads();   // doubles as P-bounce visibility + V visibility barrier

    v16bf pf0 = frag_cat(*(const v8bf*)&Pb[w][col][hf * 8],
                         *(const v8bf*)&Pb[w][col][16 + hf * 8]);
    v16bf pf1 = frag_cat(*(const v8bf*)&Pb[w][col][32 + hf * 8],
                         *(const v8bf*)&Pb[w][col][48 + hf * 8]);
#pragma unroll
    for (int f = 0; f < 4; ++f) {
      v16bf vf0 = frag_cat(*(const v8bf*)&Vs[cur][f * 16 + col][hf * 8],
                           *(const v8bf*)&Vs[cur][f * 16 + col][16 + hf * 8]);
      v16bf vf1 = frag_cat(*(const v8bf*)&Vs[cur][f * 16 + col][32 + hf * 8],
                           *(const v8bf*)&Vs[cur][f * 16 + col][48 + hf * 8]);
      Of[f] = WMMA_BF16(pf0, vf0, Of[f]);
      Of[f] = WMMA_BF16(pf1, vf1, Of[f]);
    }
  }

  // Normalize rows, store O bf16 [b, l, h*64+d] for the FC GEMM.
#pragma unroll
  for (int r = 0; r < 8; ++r) {
    float inv = 1.0f / rowsum[r];
    int q = qtile * 128 + w * 16 + hf * 8 + r;
    size_t base = ((size_t)b * 2048 + q) * 1024 + (size_t)h * 64;
#pragma unroll
    for (int f = 0; f < 4; ++f)
      Ob[base + f * 16 + col] = (bf16)(Of[f][r] * inv);
  }
}

// ---------------------------------------------------------------------------
// Host-side launcher
// ---------------------------------------------------------------------------
extern "C" void kernel_launch(void* const* d_in, const int* in_sizes, int n_in,
                              void* d_out, int out_size, void* d_ws, size_t ws_size,
                              hipStream_t stream) {
  (void)in_sizes; (void)n_in; (void)out_size; (void)ws_size;
  const float* X    = (const float*)d_in[0];
  const float* Wqkv = (const float*)d_in[1];
  const float* bqkv = (const float*)d_in[2];
  const float* Wfc  = (const float*)d_in[3];
  const float* bfc  = (const float*)d_in[4];
  float* out = (float*)d_out;

  const int Bn = 4, L = 2048, E = 1024, H = 16, DK = 64;
  const int M = Bn * L;            // 8192
  const int Nqkv = 3 * H * DK;     // 3072
  const int Nfc = E;               // 1024

  char* ws = (char*)d_ws;
  bf16* Xb  = (bf16*)ws; ws += (size_t)M * E * 2;
  bf16* Wqt = (bf16*)ws; ws += (size_t)Nqkv * E * 2;   // [3072][1024]
  bf16* Wft = (bf16*)ws; ws += (size_t)Nfc * E * 2;    // [1024][1024]
  bf16* Qbf = (bf16*)ws; ws += (size_t)Bn * H * L * DK * 2;
  bf16* Kbf = (bf16*)ws; ws += (size_t)Bn * H * L * DK * 2;
  bf16* Vbf = (bf16*)ws; ws += (size_t)Bn * H * L * DK * 2;  // [b,h,d,l]
  bf16* Obf = (bf16*)ws; ws += (size_t)M * E * 2;

  // 1) conversions: X straight; weights transpose-converted to [N][K]
  {
    int n = M * E;
    f32_to_bf16_kernel<<<(n / 4 + 255) / 256, 256, 0, stream>>>(X, Xb, n);
    transpose_f32_to_bf16_kernel<<<dim3(Nqkv / 32, E / 32), dim3(32, 8), 0, stream>>>(
        Wqkv, Wqt, E, Nqkv);
    transpose_f32_to_bf16_kernel<<<dim3(Nfc / 32, E / 32), dim3(32, 8), 0, stream>>>(
        Wfc, Wft, E, Nfc);
  }

  // 2) QKV projection GEMM + scatter epilogue (Q scaled, V head-transposed)
  gemm_bf16_kernel<0><<<dim3(Nqkv / 128, M / 128), 256, 0, stream>>>(
      Xb, Wqt, bqkv, nullptr, Qbf, Kbf, Vbf, Nqkv, E);

  // 3) flash attention
  attn_kernel<<<dim3(L / 128, Bn * H), 256, 0, stream>>>(Qbf, Kbf, Vbf, Obf);

  // 4) output projection GEMM -> f32 d_out
  gemm_bf16_kernel<1><<<dim3(Nfc / 128, M / 128), 256, 0, stream>>>(
      Obf, Wft, bfc, out, nullptr, nullptr, nullptr, Nfc, H * DK);
}
